// AFTFullAttention_36060545417317
// MI455X (gfx1250) — matrix operand order repaired and checked by reading, not compiled, not run on hardware
//
#include <hip/hip_runtime.h>

// ---------------------------------------------------------------------------
// AFT-full attention for MI455X (gfx1250), wave32 + WMMA f16->f32.
// Shapes: B=4, T=2048, DIM=1024, H=8, HD=128.
// Pipeline:
//   0) ew  = exp(wbias) -> f16   (one elementwise pass, strips exp/cvt from hot loop)
//   1) proj: Q/K/V GEMMs, epilogues sigmoid(q), exp(k) stored [b,h,d,t], v
//   2) attn: numer GEMM over all 4 batches/wave, LDS-staged A tile via
//      GLOBAL_LOAD_ASYNC_TO_LDS (ASYNCcnt); fused batch reduction
//      -> y = sigmoid(q) * (sum_b numer*v)/(sum_b numer)
//   3) out: y @ Wo^T + bo
// ---------------------------------------------------------------------------

typedef _Float16 v16h __attribute__((ext_vector_type(16)));
typedef _Float16 v8h  __attribute__((ext_vector_type(8)));
typedef _Float16 v4h  __attribute__((ext_vector_type(4)));
typedef float    v8f  __attribute__((ext_vector_type(8)));

#define B_    4
#define T_    2048
#define DIM_  1024
#define H_    8
#define HD_   128
#define BT_   (B_ * T_)            // 8192
#define BSTR_ (T_ * DIM_)          // 2097152  (per-batch flat stride)
#define HSTR_ (T_ * HD_)           // 262144   (per-head flat stride)

#if __has_builtin(__builtin_amdgcn_global_load_async_to_lds_b32) && \
    __has_builtin(__builtin_amdgcn_s_wait_asynccnt)
#define USE_ASYNC_LDS 1
#else
#define USE_ASYNC_LDS 0
#endif

#if USE_ASYNC_LDS
typedef __attribute__((address_space(1))) int* gas1_int;   // global int*
typedef __attribute__((address_space(3))) int* las3_int;   // LDS int*
#endif

// WMMA wrapper: D = A(16x32 f16) * B(32x16 f16) + C(16x16 f32)
static __device__ __forceinline__ v8f wmma_f16(v16h a, v16h b, v8f c) {
    return __builtin_amdgcn_wmma_f32_16x16x32_f16(
        /*neg_a=*/false, a, /*neg_b=*/false, b,
        /*c_mod=*/(short)0, c, /*reuse_a=*/false, /*reuse_b=*/false);
}

// Build v16h from 16 consecutive f16 in memory (32B, two b128 loads).
static __device__ __forceinline__ v16h load_h16(const _Float16* p) {
    v8h x0 = *(const v8h*)p;
    v8h x1 = *(const v8h*)(p + 8);
    v16h r;
#pragma unroll
    for (int i = 0; i < 8; ++i) { r[i] = x0[i]; r[8 + i] = x1[i]; }
    return r;
}

// Build v16h (A-operand chunks) from 2x8 consecutive f32, converting to f16.
static __device__ __forceinline__ v16h cvt_a_f32(const float* p0, const float* p1) {
    float4 q0 = *(const float4*)(p0);
    float4 q1 = *(const float4*)(p0 + 4);
    float4 q2 = *(const float4*)(p1);
    float4 q3 = *(const float4*)(p1 + 4);
    v16h a;
    a[0]  = (_Float16)q0.x; a[1]  = (_Float16)q0.y; a[2]  = (_Float16)q0.z; a[3]  = (_Float16)q0.w;
    a[4]  = (_Float16)q1.x; a[5]  = (_Float16)q1.y; a[6]  = (_Float16)q1.z; a[7]  = (_Float16)q1.w;
    a[8]  = (_Float16)q2.x; a[9]  = (_Float16)q2.y; a[10] = (_Float16)q2.z; a[11] = (_Float16)q2.w;
    a[12] = (_Float16)q3.x; a[13] = (_Float16)q3.y; a[14] = (_Float16)q3.z; a[15] = (_Float16)q3.w;
    return a;
}

// Build v16h (B-operand) from 16 consecutive f32, converting to f16.
static __device__ __forceinline__ v16h cvt_b_f32(const float* p) {
    float4 q0 = *(const float4*)(p);
    float4 q1 = *(const float4*)(p + 4);
    float4 q2 = *(const float4*)(p + 8);
    float4 q3 = *(const float4*)(p + 12);
    v16h b;
    b[0]  = (_Float16)q0.x; b[1]  = (_Float16)q0.y; b[2]  = (_Float16)q0.z; b[3]  = (_Float16)q0.w;
    b[4]  = (_Float16)q1.x; b[5]  = (_Float16)q1.y; b[6]  = (_Float16)q1.z; b[7]  = (_Float16)q1.w;
    b[8]  = (_Float16)q2.x; b[9]  = (_Float16)q2.y; b[10] = (_Float16)q2.z; b[11] = (_Float16)q2.w;
    b[12] = (_Float16)q3.x; b[13] = (_Float16)q3.y; b[14] = (_Float16)q3.z; b[15] = (_Float16)q3.w;
    return b;
}

// ---------------------------------------------------------------------------
// Kernel 0: ew = exp(wbias) as f16, vectorized x4.  33.5M elements.
// ---------------------------------------------------------------------------
__global__ __launch_bounds__(256) void ew_kernel(
    const float* __restrict__ wbias, _Float16* __restrict__ ew)
{
    const size_t i4 = ((size_t)blockIdx.x * 256 + threadIdx.x) * 4;
    float4 q = *(const float4*)(wbias + i4);
    v4h o;
    o[0] = (_Float16)__expf(q.x);
    o[1] = (_Float16)__expf(q.y);
    o[2] = (_Float16)__expf(q.z);
    o[3] = (_Float16)__expf(q.w);
    *(v4h*)(ew + i4) = o;
}

// ---------------------------------------------------------------------------
// Kernel 1: fused QKV projection.  out[r,c] = sum_k x[r,k] * W[c,k] + bias[c]
// One wave = one 16x16 output tile.  blockIdx.y selects Q(0)/K(1)/V(2).
// ---------------------------------------------------------------------------
__global__ __launch_bounds__(256) void proj_kernel(
    const float* __restrict__ x,
    const float* __restrict__ Wq, const float* __restrict__ bq,
    const float* __restrict__ Wk, const float* __restrict__ bk,
    const float* __restrict__ Wv, const float* __restrict__ bv,
    _Float16* __restrict__ sq, _Float16* __restrict__ ekT,
    _Float16* __restrict__ vv)
{
    const int mode = blockIdx.y;
    const float* W    = (mode == 0) ? Wq : (mode == 1) ? Wk : Wv;
    const float* bias = (mode == 0) ? bq : (mode == 1) ? bk : bv;

    const int lane = threadIdx.x & 31;
    const int wave = threadIdx.x >> 5;
    const int tile = blockIdx.x * 8 + wave;        // 0..32767
    const int row0 = (tile >> 6) * 16;             // 512 row tiles
    const int col0 = (tile & 63) * 16;             // 64  col tiles
    const int half = lane >> 4;
    const int mn   = lane & 15;

    const float* arow = x + (size_t)(row0 + mn) * DIM_;
    const float* brow = W + (size_t)(col0 + mn) * DIM_;

    v8f acc = {};
#pragma unroll 4
    for (int k0 = 0; k0 < DIM_; k0 += 32) {
        v16h a = cvt_a_f32(arow + k0 + half * 16, arow + k0 + half * 16 + 8);
        v16h b = cvt_b_f32(brow + k0 + half * 16);
        acc = wmma_f16(a, b, acc);
        if (k0 + 32 < DIM_) {                      // global_prefetch_b8
            __builtin_prefetch(arow + k0 + 32, 0, 0);
            __builtin_prefetch(brow + k0 + 32, 0, 0);
        }
    }

    const float bval = bias[col0 + mn];
#pragma unroll
    for (int r = 0; r < 8; ++r) {
        const int grow = row0 + r + 8 * half;      // global row (b*T + t')
        const int gcol = col0 + mn;                // global col (c' in H*HD)
        const float val = acc[r] + bval;
        if (mode == 0) {
            sq[(size_t)grow * DIM_ + gcol] = (_Float16)(1.0f / (1.0f + __expf(-val)));
        } else if (mode == 2) {
            vv[(size_t)grow * DIM_ + gcol] = (_Float16)val;
        } else {
            // flat reinterpretation [B,T,H*HD] -> [B,H,T,HD], then transpose t<->d
            const int bb   = grow >> 11;           // / T_
            const int flat = (grow & (T_ - 1)) * DIM_ + gcol;
            const int h    = flat >> 18;           // / HSTR_
            const int rem  = flat & (HSTR_ - 1);
            const int t    = rem >> 7;             // / HD_
            const int d    = rem & (HD_ - 1);
            ekT[(size_t)bb * BSTR_ + (size_t)h * HSTR_ + (size_t)d * T_ + t] =
                (_Float16)__expf(val);
        }
    }
}

// ---------------------------------------------------------------------------
// Kernel 2: numer[b,h,t,d] = sum_s ew[h,t,s] * ek[b,h,s,d], fused batch
// reduction + sigmoid gating.  Block = 8 waves (wave = d-tile), A tile shared
// through LDS; staged with GLOBAL_LOAD_ASYNC_TO_LDS (ASYNCcnt).
// ---------------------------------------------------------------------------
__global__ __launch_bounds__(256) void attn_kernel(
    const _Float16* __restrict__ ew,
    const _Float16* __restrict__ ekT,
    const _Float16* __restrict__ sq,
    const _Float16* __restrict__ vv,
    _Float16*       __restrict__ y)
{
    __shared__ __align__(16) _Float16 lds_a[16 * 32];   // 1 KB tile

    const int t0   = blockIdx.x * 16;              // 128 t-tiles
    const int h    = blockIdx.y;                   // 8 heads
    const int lane = threadIdx.x & 31;
    const int wave = threadIdx.x >> 5;             // d-tile 0..7
    const int d0   = wave * 16;
    const int half = lane >> 4;
    const int mn   = lane & 15;

    const _Float16* ewrow = ew + (size_t)h * T_ * T_ + (size_t)t0 * T_;

    // B operand base: ekT[b, h, d0+mn, s] — s contiguous
    const _Float16* bp = ekT + (size_t)h * HSTR_ + (size_t)(d0 + mn) * T_;

    // Per-thread dword slot of the 16x32 f16 tile (256 dwords total)
    const int e    = threadIdx.x;
    const int srow = e >> 4;             // 0..15
    const int scol = (e & 15) * 2;       // 0,2,..,30

    v8f acc0 = {}, acc1 = {}, acc2 = {}, acc3 = {};

    for (int s0 = 0; s0 < T_; s0 += 32) {
        const _Float16* src = ewrow + (size_t)srow * T_ + s0 + scol;
#if USE_ASYNC_LDS
        __builtin_amdgcn_global_load_async_to_lds_b32(
            (gas1_int)(void*)src,
            (las3_int)(void*)&lds_a[e * 2],
            /*offset=*/0, /*cpol=*/0);
        __builtin_amdgcn_s_wait_asynccnt(0);
#else
        ((unsigned int*)lds_a)[e] = *(const unsigned int*)src;
#endif
        __syncthreads();

        // A operand from LDS (ds_read_b128 x2 per lane)
        const _Float16* la = &lds_a[mn * 32 + half * 16];
        v8h ca0 = *(const v8h*)la;
        v8h ca1 = *(const v8h*)(la + 8);
        v16h a;
#pragma unroll
        for (int i = 0; i < 8; ++i) { a[i] = ca0[i]; a[8 + i] = ca1[i]; }

        const int soff = s0 + half * 16;
        v16h b0 = load_h16(bp + 0 * BSTR_ + soff);
        v16h b1 = load_h16(bp + 1 * BSTR_ + soff);
        v16h b2 = load_h16(bp + 2 * BSTR_ + soff);
        v16h b3 = load_h16(bp + 3 * BSTR_ + soff);

        acc0 = wmma_f16(a, b0, acc0);
        acc1 = wmma_f16(a, b1, acc1);
        acc2 = wmma_f16(a, b2, acc2);
        acc3 = wmma_f16(a, b3, acc3);

        if (s0 + 32 < T_) {
            __builtin_prefetch(src + 32, 0, 0);
            __builtin_prefetch(bp + s0 + 32, 0, 0);
        }
        __syncthreads();   // protect lds_a before next fill
    }

    // Epilogue: batch reduction + gating, straight from accumulators.
#pragma unroll
    for (int r = 0; r < 8; ++r) {
        const int t = t0 + r + 8 * half;
        const size_t off = (size_t)h * HSTR_ + (size_t)t * HD_ + (d0 + mn);
        const float n0 = acc0[r], n1 = acc1[r], n2 = acc2[r], n3 = acc3[r];
        const float w0 = (float)vv[off + 0 * BSTR_];
        const float w1 = (float)vv[off + 1 * BSTR_];
        const float w2 = (float)vv[off + 2 * BSTR_];
        const float w3 = (float)vv[off + 3 * BSTR_];
        const float denom = n0 + n1 + n2 + n3;                  // > 0 (exp sums)
        const float wgt   = (n0 * w0 + n1 * w1 + n2 * w2 + n3 * w3) / denom;
        y[off + 0 * BSTR_] = (_Float16)((float)sq[off + 0 * BSTR_] * wgt);
        y[off + 1 * BSTR_] = (_Float16)((float)sq[off + 1 * BSTR_] * wgt);
        y[off + 2 * BSTR_] = (_Float16)((float)sq[off + 2 * BSTR_] * wgt);
        y[off + 3 * BSTR_] = (_Float16)((float)sq[off + 3 * BSTR_] * wgt);
    }
}

// ---------------------------------------------------------------------------
// Kernel 3: out[r,c] = sum_k y[r,k] * Wo[c,k] + bo[c]
// ---------------------------------------------------------------------------
__global__ __launch_bounds__(256) void out_kernel(
    const _Float16* __restrict__ y, const float* __restrict__ Wo,
    const float* __restrict__ bo, float* __restrict__ out)
{
    const int lane = threadIdx.x & 31;
    const int wave = threadIdx.x >> 5;
    const int tile = blockIdx.x * 8 + wave;
    const int row0 = (tile >> 6) * 16;
    const int col0 = (tile & 63) * 16;
    const int half = lane >> 4;
    const int mn   = lane & 15;

    const _Float16* arow = y  + (size_t)(row0 + mn) * DIM_;
    const float*    brow = Wo + (size_t)(col0 + mn) * DIM_;

    v8f acc = {};
#pragma unroll 4
    for (int k0 = 0; k0 < DIM_; k0 += 32) {
        const _Float16* pa = arow + k0 + half * 16;
        v8h c0 = *(const v8h*)pa;
        v8h c1 = *(const v8h*)(pa + 8);
        v16h a;
#pragma unroll
        for (int i = 0; i < 8; ++i) { a[i] = c0[i]; a[8 + i] = c1[i]; }
        v16h b = cvt_b_f32(brow + k0 + half * 16);
        acc = wmma_f16(a, b, acc);
        if (k0 + 32 < DIM_) {
            __builtin_prefetch(arow + k0 + 32, 0, 0);
            __builtin_prefetch(brow + k0 + 32, 0, 0);
        }
    }

    const float bval = bo[col0 + mn];
#pragma unroll
    for (int r = 0; r < 8; ++r) {
        const int grow = row0 + r + 8 * half;
        out[(size_t)grow * DIM_ + (col0 + mn)] = acc[r] + bval;
    }
}

// ---------------------------------------------------------------------------
extern "C" void kernel_launch(void* const* d_in, const int* in_sizes, int n_in,
                              void* d_out, int out_size, void* d_ws, size_t ws_size,
                              hipStream_t stream) {
    (void)in_sizes; (void)n_in; (void)out_size; (void)ws_size;
    const float* x     = (const float*)d_in[0];
    const float* Wq    = (const float*)d_in[1];
    const float* bq    = (const float*)d_in[2];
    const float* Wk    = (const float*)d_in[3];
    const float* bk    = (const float*)d_in[4];
    const float* Wv    = (const float*)d_in[5];
    const float* bv    = (const float*)d_in[6];
    const float* wbias = (const float*)d_in[7];
    const float* Wo    = (const float*)d_in[8];
    const float* bo    = (const float*)d_in[9];
    float* out = (float*)d_out;

    // Workspace layout (f16): sq | ekT | vv | y  (16 MB each) | ew (64 MB)
    _Float16* sq  = (_Float16*)d_ws;
    _Float16* ekT = sq  + (size_t)BT_ * DIM_;
    _Float16* vv  = ekT + (size_t)BT_ * DIM_;
    _Float16* yb  = vv  + (size_t)BT_ * DIM_;
    _Float16* ew  = yb  + (size_t)BT_ * DIM_;

    ew_kernel<<<dim3((H_ * T_ * T_) / (256 * 4)), 256, 0, stream>>>(wbias, ew);
    proj_kernel<<<dim3(4096, 3), 256, 0, stream>>>(x, Wq, bq, Wk, bk, Wv, bv,
                                                   sq, ekT, vv);
    attn_kernel<<<dim3(128, 8), 256, 0, stream>>>(ew, ekT, sq, vv, yb);
    out_kernel<<<dim3(4096), 256, 0, stream>>>(yb, Wo, bo, out);
}
